// GuoCapSAREncoder_64141041599136
// MI455X (gfx1250) — matrix operand order repaired and checked by reading, not compiled, not run on hardware
//
#include <hip/hip_runtime.h>
#include <math.h>

// ---------------------------------------------------------------------------
// Types for CDNA5 WMMA (wave32)
// ---------------------------------------------------------------------------
typedef __bf16 bf16;
typedef __attribute__((ext_vector_type(16))) __bf16 bf16x16;
typedef __attribute__((ext_vector_type(8)))  __bf16 bf16x8;
typedef __attribute__((ext_vector_type(8)))  float  floatx8;
typedef __attribute__((ext_vector_type(4)))  unsigned int u32x4;
typedef __attribute__((ext_vector_type(8)))  int i32x8;
typedef __attribute__((ext_vector_type(4)))  int i32x4;

#define EPS 1e-8f

// ---------------------------------------------------------------------------
// Tensor Data Mover: 2D tile load Global -> LDS.
// All size quantities in 8-byte units (D#.data_size = 3 -> 8B elements).
//   tile0   : tile width  (row length to copy)        [16-bit field]
//   tile1   : tile height (#rows)                     [16-bit field]
//   stride0 : tensor_dim0_stride (row pitch)          [48-bit field]
//   tdim0/1 : tensor dims for OOB zero-fill
// Issue from ONE wave only; track with TENSORcnt.
// ---------------------------------------------------------------------------
__device__ __forceinline__ void tdm_load_2d(void* lds, const void* g,
                                            unsigned tile0, unsigned tile1,
                                            unsigned long long stride0,
                                            unsigned tdim0, unsigned tdim1) {
    unsigned long long ga = (unsigned long long)(size_t)g;
    u32x4 g0;
    g0[0] = 1u;                                          // count=1, user mode
    g0[1] = (unsigned)(size_t)lds;                       // lds_addr (low 32 of generic)
    g0[2] = (unsigned)ga;                                // global_addr[31:0]
    g0[3] = (unsigned)((ga >> 32) & 0x01ffffffull)       // global_addr[56:32]
          | (2u << 30);                                  // type=2 ("image")
    // group1 (256 bits), assembled as four 64-bit lanes
    unsigned long long q0 = (3ull << 16)                                  // data_size=8B
                          | ((unsigned long long)(tdim0 & 0xffffu) << 48); // tensor_dim0[15:0] @48
    unsigned long long q1 = (unsigned long long)(tdim0 >> 16)             // tensor_dim0[31:16]
                          | ((unsigned long long)tdim1 << 16)             // tensor_dim1 @80
                          | ((unsigned long long)(tile0 & 0xffffu) << 48);// tile_dim0 @112
    unsigned long long q2 = (unsigned long long)(tile1 & 0xffffu)         // tile_dim1 @128
                          | ((stride0 & 0xffffffffull) << 32);            // stride0[31:0] @160
    unsigned long long q3 = (stride0 >> 32) & 0xffffull;                  // stride0[47:32] @192
    i32x8 g1;
    g1[0] = (int)(unsigned)q0; g1[1] = (int)(unsigned)(q0 >> 32);
    g1[2] = (int)(unsigned)q1; g1[3] = (int)(unsigned)(q1 >> 32);
    g1[4] = (int)(unsigned)q2; g1[5] = (int)(unsigned)(q2 >> 32);
    g1[6] = (int)(unsigned)q3; g1[7] = (int)(unsigned)(q3 >> 32);
    i32x4 z4 = {};
#if __clang_major__ >= 23
    i32x8 z8 = {};
    __builtin_amdgcn_tensor_load_to_lds(g0, g1, z4, z4, z8, 0);
#else
    __builtin_amdgcn_tensor_load_to_lds(g0, g1, z4, z4, 0);
#endif
}

#define LD8(p)  (*reinterpret_cast<const bf16x8*>(p))
#define CAT16(lo, hi) __builtin_shufflevector(lo, hi, 0,1,2,3,4,5,6,7,8,9,10,11,12,13,14,15)

// ---------------------------------------------------------------------------
// bf16 GEMM:  C[M,N] = A[M,K] * Bt[N,K]^T (+bias)    (K % 32 == 0)
// Block tile 128x64, 8 waves, wave tile 32x32 (4 WMMAs / K-step).
// TDM double-buffered pipeline: wave 0 issues tensor_load_to_lds for the
// next K tile while all waves compute the current one.
// mode: 0 = f32 out, 1 = bf16 out, 2 = bf16 out + ReLU.
// ---------------------------------------------------------------------------
__global__ __launch_bounds__(256)
void gemm_wmma_bf16(const bf16* __restrict__ A, const bf16* __restrict__ Bt,
                    void* __restrict__ Cout, const float* __restrict__ bias,
                    int M, int N, int K, int mode) {
    __shared__ __align__(16) bf16 As[2][128 * 32];
    __shared__ __align__(16) bf16 Bs[2][64 * 32];   // [n][k]

    const int tid  = threadIdx.x;
    const int lane = tid & 31;
    const int wave = tid >> 5;
    const int half = lane >> 4;
    const int l15  = lane & 15;
    const int wm   = (wave >> 1) * 32;    // wave M origin in block tile
    const int wn   = (wave & 1) * 32;     // wave N origin in block tile
    const int mblk = blockIdx.y * 128;
    const int nblk = blockIdx.x * 64;
    const int nsteps = K >> 5;
    const unsigned long long rowpitch8 = (unsigned long long)(K >> 2); // K*2B / 8

    floatx8 acc00 = {}, acc01 = {}, acc10 = {}, acc11 = {};
    const bool issuer = (wave == 0);

    if (issuer) {
        tdm_load_2d(&As[0][0], A + (size_t)mblk * K, 8, 128, rowpitch8,
                    8, (unsigned)(M - mblk));
        tdm_load_2d(&Bs[0][0], Bt + (size_t)nblk * K, 8, 64, rowpitch8,
                    8, (unsigned)(N - nblk));
    }

    for (int s = 0; s < nsteps; ++s) {
        const int cur = s & 1;
        if (issuer) __builtin_amdgcn_s_wait_tensorcnt(0);
        __syncthreads();                       // current buffer ready for all waves
        if (issuer && (s + 1) < nsteps) {      // prefetch next K tile into other buffer
            const int k0 = (s + 1) * 32;
            tdm_load_2d(&As[cur ^ 1][0], A + (size_t)mblk * K + k0, 8, 128, rowpitch8,
                        8, (unsigned)(M - mblk));
            tdm_load_2d(&Bs[cur ^ 1][0], Bt + (size_t)nblk * K + k0, 8, 64, rowpitch8,
                        8, (unsigned)(N - nblk));
        }
        const bf16* as = &As[cur][0];
        const bf16* bs = &Bs[cur][0];
        // A fragments (16x32): elems 0..7 -> K = half*8+0..7, 8..15 -> K = 16+half*8+0..7
        bf16x16 a0 = CAT16(LD8(&as[(wm + l15) * 32 + half * 8]),
                           LD8(&as[(wm + l15) * 32 + 16 + half * 8]));
        bf16x16 a1 = CAT16(LD8(&as[(wm + 16 + l15) * 32 + half * 8]),
                           LD8(&as[(wm + 16 + l15) * 32 + 16 + half * 8]));
        // B fragments (32x16, LDS [n][k]): elems 0..7 -> K = half*8+0..7, 8..15 -> K = 16+half*8+0..7
        bf16x16 b0 = CAT16(LD8(&bs[(wn + l15) * 32 + half * 8]),
                           LD8(&bs[(wn + l15) * 32 + 16 + half * 8]));
        bf16x16 b1 = CAT16(LD8(&bs[(wn + 16 + l15) * 32 + half * 8]),
                           LD8(&bs[(wn + 16 + l15) * 32 + 16 + half * 8]));
        acc00 = __builtin_amdgcn_wmma_f32_16x16x32_bf16(false, a0, false, b0, (short)0, acc00, false, false);
        acc01 = __builtin_amdgcn_wmma_f32_16x16x32_bf16(false, a0, false, b1, (short)0, acc01, false, false);
        acc10 = __builtin_amdgcn_wmma_f32_16x16x32_bf16(false, a1, false, b0, (short)0, acc10, false, false);
        acc11 = __builtin_amdgcn_wmma_f32_16x16x32_bf16(false, a1, false, b1, (short)0, acc11, false, false);
        // top-of-loop barrier orders these reads against next TDM overwrite
    }

    // epilogue: tile (ti,tj) -> rows mblk+wm+ti*16+half*8+r, col nblk+wn+tj*16+l15
    const int gn0 = nblk + wn + l15;
    const int gn1 = nblk + wn + 16 + l15;
    const float bv0 = (bias && gn0 < N) ? bias[gn0] : 0.0f;
    const float bv1 = (bias && gn1 < N) ? bias[gn1] : 0.0f;
#pragma unroll
    for (int r = 0; r < 8; ++r) {
        const int gmA = mblk + wm + half * 8 + r;
        const int gmB = gmA + 16;
        float v;
#define STORE_ELEM(gm, gn, bv, accv)                                        \
        if ((gm) < M && (gn) < N) {                                         \
            v = (accv)[r] + (bv);                                           \
            if (mode == 2) v = fmaxf(v, 0.0f);                              \
            if (mode == 0) ((float*)Cout)[(size_t)(gm) * N + (gn)] = v;     \
            else           ((bf16*)Cout)[(size_t)(gm) * N + (gn)] = (bf16)v;\
        }
        STORE_ELEM(gmA, gn0, bv0, acc00)
        STORE_ELEM(gmA, gn1, bv1, acc01)
        STORE_ELEM(gmB, gn0, bv0, acc10)
        STORE_ELEM(gmB, gn1, bv1, acc11)
#undef STORE_ELEM
    }
}

// ---------------------------------------------------------------------------
// im2col kernels (bf16 outputs, K padded to multiple of 32 where needed)
// ---------------------------------------------------------------------------
__global__ __launch_bounds__(256)
void im2col1(const float* __restrict__ x, bf16* __restrict__ A, int b0) {
    const size_t total = (size_t)8 * 7056 * 96;
    size_t idx = (size_t)blockIdx.x * 256 + threadIdx.x;
    if (idx >= total) return;
    int k  = (int)(idx % 96);
    int ml = (int)(idx / 96);
    float v = 0.0f;
    if (k < 81) {
        int kh = k / 9, kw = k % 9;
        int bl = ml / 7056, pix = ml % 7056;
        int oh = pix / 84, ow = pix % 84;
        v = x[((size_t)(b0 + bl) * 92 + (oh + kh)) * 92 + (ow + kw)];
    }
    A[idx] = (bf16)v;
}

__global__ __launch_bounds__(256)
void im2col2(const bf16* __restrict__ pooled, bf16* __restrict__ A, int b0) {
    const size_t total = (size_t)8 * 1444 * 3200;
    size_t idx = (size_t)blockIdx.x * 256 + threadIdx.x;
    if (idx >= total) return;
    int k  = (int)(idx % 3200);
    int ml = (int)(idx / 3200);
    int c = k & 127, r = k >> 7;
    int kh = r / 5, kw = r % 5;
    int bl = ml / 1444, pix = ml % 1444;
    int oh = pix / 38, ow = pix % 38;
    A[idx] = pooled[(((size_t)(b0 + bl) * 42 + (oh + kh)) * 42 + (ow + kw)) * 128 + c];
}

__global__ __launch_bounds__(256)
void im2col3(const bf16* __restrict__ cin, bf16* __restrict__ A, int b0) {
    const size_t total = (size_t)8 * 256 * 4096;
    size_t idx = (size_t)blockIdx.x * 256 + threadIdx.x;
    if (idx >= total) return;
    int k  = (int)(idx % 4096);
    int ml = (int)(idx / 4096);
    int c = k & 63, r = k >> 6;
    int kh = r >> 3, kw = r & 7;
    int bl = ml >> 8, pix = ml & 255;
    int oh = pix >> 4, ow = pix & 15;
    int ih = oh * 2 + kh, iw = ow * 2 + kw;
    A[idx] = cin[(((size_t)(b0 + bl) * 38 + ih) * 38 + iw) * 64 + c];
}

__global__ __launch_bounds__(256)
void maxpool2(const bf16* __restrict__ cin, bf16* __restrict__ pooled, int b0) {
    const size_t total = (size_t)8 * 42 * 42 * 128;
    size_t idx = (size_t)blockIdx.x * 256 + threadIdx.x;
    if (idx >= total) return;
    int c = (int)(idx & 127);
    int r = (int)(idx >> 7);
    int w = r % 42; r /= 42;
    int h = r % 42; int bl = r / 42;
    const size_t base = (size_t)bl * 84;
    float m0 = (float)cin[((base + 2*h    ) * 84 + 2*w    ) * 128 + c];
    float m1 = (float)cin[((base + 2*h    ) * 84 + 2*w + 1) * 128 + c];
    float m2 = (float)cin[((base + 2*h + 1) * 84 + 2*w    ) * 128 + c];
    float m3 = (float)cin[((base + 2*h + 1) * 84 + 2*w + 1) * 128 + c];
    float m = fmaxf(fmaxf(m0, m1), fmaxf(m2, m3));
    pooled[(((size_t)(b0 + bl) * 42 + h) * 42 + w) * 128 + c] = (bf16)m;
}

// ---------------------------------------------------------------------------
// Weight prep: produce Bt[n][k] bf16 (K padded where needed)
// ---------------------------------------------------------------------------
__global__ __launch_bounds__(256)
void prepB1t(const float* __restrict__ w1, bf16* __restrict__ Bt) { // [128][96]
    int idx = blockIdx.x * 256 + threadIdx.x;
    if (idx >= 128 * 96) return;
    int k = idx % 96, o = idx / 96;
    Bt[idx] = (bf16)((k < 81) ? w1[o * 81 + k] : 0.0f);
}
__global__ __launch_bounds__(256)
void prepB2t(const float* __restrict__ w2, bf16* __restrict__ Bt) { // [64][3200]
    int idx = blockIdx.x * 256 + threadIdx.x;
    if (idx >= 64 * 3200) return;
    int k = idx % 3200, o = idx / 3200;
    int c = k & 127, r = k >> 7;
    int kh = r / 5, kw = r % 5;
    Bt[idx] = (bf16)w2[(((size_t)o * 128 + c) * 5 + kh) * 5 + kw];
}
__global__ __launch_bounds__(256)
void prepBct(const float* __restrict__ wc, bf16* __restrict__ Bt) { // [128][4096]
    int idx = blockIdx.x * 256 + threadIdx.x;
    if (idx >= 128 * 4096) return;
    int k = idx % 4096, o = idx / 4096;
    int c = k & 63, r = k >> 6;
    int kh = r >> 3, kw = r & 7;
    Bt[idx] = (bf16)wc[(((size_t)o * 64 + c) * 8 + kh) * 8 + kw];
}
__global__ __launch_bounds__(256)
void prepWrBf(const float* __restrict__ Wr, bf16* __restrict__ Wb) { // 1024*160*32
    size_t idx = (size_t)blockIdx.x * 256 + threadIdx.x;
    if (idx >= (size_t)1024 * 160 * 32) return;
    Wb[idx] = (bf16)Wr[idx];
}
__global__ __launch_bounds__(256)
void zero_f32(float* __restrict__ p, int n) {
    int idx = blockIdx.x * 256 + threadIdx.x;
    if (idx < n) p[idx] = 0.0f;
}

// ---------------------------------------------------------------------------
// Squash primary capsules: caps f32 NHWC [64*256,128] -> u bf16 [64,1024,32]
// ---------------------------------------------------------------------------
__global__ __launch_bounds__(256)
void squash_caps(const float* __restrict__ caps, bf16* __restrict__ u) {
    int gw   = blockIdx.x * 8 + (threadIdx.x >> 5);
    int lane = threadIdx.x & 31;
    int b = gw >> 10, pp = gw & 1023;
    int ct = pp >> 8, pix = pp & 255;
    float val = caps[((size_t)b * 256 + pix) * 128 + ct * 32 + lane];
    float n2 = val * val;
#pragma unroll
    for (int off = 16; off; off >>= 1) n2 += __shfl_xor(n2, off, 32);
    float sc = (n2 / (1.0f + n2)) * rsqrtf(n2 + EPS);
    u[((size_t)b * 1024 + pp) * 32 + lane] = (bf16)(val * sc);
}

// ---------------------------------------------------------------------------
// u_hat: per primary capsule p, [64x32] x [32x160] -> [64x160] via WMMA.
// LDS staged via two TDM loads (strided 2D for u, flat for Wr[p]).
// ---------------------------------------------------------------------------
__global__ __launch_bounds__(256)
void uhat_wmma(const bf16* __restrict__ u, const bf16* __restrict__ Wb,
               float* __restrict__ uhat) {
    const int p = blockIdx.x;
    __shared__ __align__(16) bf16 Us[64 * 32];
    __shared__ __align__(16) bf16 Ws[160 * 32];
    const int tid = threadIdx.x;

    if (tid < 32) {
        // u rows: 64 rows of 32 bf16 (8 units), row pitch 1024*32 bf16 = 8192 units
        tdm_load_2d(&Us[0], u + (size_t)p * 32, 8, 64, 8192ull, 8, 64);
        // Wr[p]: contiguous 160*32 bf16 = 1280 units, 1 row
        tdm_load_2d(&Ws[0], Wb + (size_t)p * 5120, 1280, 1, 1280ull, 1280, 1);
        __builtin_amdgcn_s_wait_tensorcnt(0);
    }
    __syncthreads();

    const int lane = tid & 31, wave = tid >> 5;
    const int half = lane >> 4, l15 = lane & 15;
    for (int t = wave; t < 40; t += 8) {
        const int m0 = (t / 10) * 16;
        const int n0 = (t % 10) * 16;
        bf16x16 afrag = CAT16(LD8(&Us[(m0 + l15) * 32 + half * 8]),
                              LD8(&Us[(m0 + l15) * 32 + 16 + half * 8]));
        bf16x16 bfrag = CAT16(LD8(&Ws[(n0 + l15) * 32 + half * 8]),
                              LD8(&Ws[(n0 + l15) * 32 + 16 + half * 8]));
        floatx8 acc = {};
        acc = __builtin_amdgcn_wmma_f32_16x16x32_bf16(false, afrag, false, bfrag,
                                                      (short)0, acc, false, false);
#pragma unroll
        for (int r = 0; r < 8; ++r) {
            int gb = m0 + half * 8 + r;     // batch index
            int gn = n0 + l15;              // t*16+o
            uhat[((size_t)gb * 1024 + p) * 160 + gn] = acc[r];
        }
    }
}

// ---------------------------------------------------------------------------
// One routing iteration (deterministic). One block per batch element.
// ---------------------------------------------------------------------------
__global__ __launch_bounds__(256)
void routing_iter(const float* __restrict__ uhat, float* __restrict__ blog,
                  float* __restrict__ out, int last) {
    const int b = blockIdx.x;
    __shared__ float sC[1024 * 10];
    __shared__ float sS[160];
    __shared__ float sV[160];
    const int tid = threadIdx.x;

    for (int p = tid; p < 1024; p += 256) {
        const float* bl = &blog[((size_t)b * 1024 + p) * 10];
        float l[10], mx = -1e30f;
#pragma unroll
        for (int t = 0; t < 10; ++t) { l[t] = bl[t]; mx = fmaxf(mx, l[t]); }
        float s = 0.0f;
#pragma unroll
        for (int t = 0; t < 10; ++t) { l[t] = __expf(l[t] - mx); s += l[t]; }
        float inv = 1.0f / s;
#pragma unroll
        for (int t = 0; t < 10; ++t) sC[p * 10 + t] = l[t] * inv;
    }
    __syncthreads();

    if (tid < 160) {
        const int t = tid >> 4;
        float acc = 0.0f;
        const float* ub = &uhat[(size_t)b * 1024 * 160];
        for (int p = 0; p < 1024; ++p)
            acc += sC[p * 10 + t] * ub[(size_t)p * 160 + tid];
        sS[tid] = acc;
    }
    __syncthreads();

    if (tid < 10) {
        float n2 = 0.0f;
#pragma unroll
        for (int o = 0; o < 16; ++o) { float sv = sS[tid * 16 + o]; n2 += sv * sv; }
        float sc = (n2 / (1.0f + n2)) * rsqrtf(n2 + EPS);
#pragma unroll
        for (int o = 0; o < 16; ++o) sV[tid * 16 + o] = sS[tid * 16 + o] * sc;
        if (last) out[b * 10 + tid] = sqrtf(n2 * sc * sc + EPS);
    }
    __syncthreads();

    if (!last) {
        for (int p = tid; p < 1024; p += 256) {
            const float* up = &uhat[((size_t)b * 1024 + p) * 160];
            float* bl = &blog[((size_t)b * 1024 + p) * 10];
#pragma unroll
            for (int t = 0; t < 10; ++t) {
                float d = 0.0f;
#pragma unroll
                for (int o = 0; o < 16; ++o) d += up[t * 16 + o] * sV[t * 16 + o];
                bl[t] += d;
            }
        }
    }
}

// ---------------------------------------------------------------------------
// Host launcher
// ---------------------------------------------------------------------------
extern "C" void kernel_launch(void* const* d_in, const int* in_sizes, int n_in,
                              void* d_out, int out_size, void* d_ws, size_t ws_size,
                              hipStream_t stream) {
    (void)in_sizes; (void)n_in; (void)out_size; (void)ws_size;
    const float* x  = (const float*)d_in[0];
    const float* w1 = (const float*)d_in[1];
    const float* b1 = (const float*)d_in[2];
    const float* w2 = (const float*)d_in[3];
    const float* b2 = (const float*)d_in[4];
    const float* wc = (const float*)d_in[5];
    const float* bc = (const float*)d_in[6];
    const float* Wr = (const float*)d_in[7];
    float* out = (float*)d_out;
    char* ws = (char*)d_ws;

    const size_t o_pooled = 0;                       // [64,42,42,128] bf16
    const size_t o_conv2  = o_pooled + 28901376;     // [64,38,38,64]  bf16
    const size_t o_caps   = o_conv2  + 11829248;     // [64,256,128]   f32
    const size_t o_u      = o_caps   + 8388608;      // [64,1024,32]   bf16
    const size_t o_uhat   = o_u      + 4194304;      // [64,1024,160]  f32
    const size_t o_blog   = o_uhat   + 41943040;     // [64,1024,10]   f32
    const size_t o_B1     = o_blog   + 2621440;      // [128,96]  bf16
    const size_t o_B2     = o_B1     + 24576;        // [64,3200] bf16
    const size_t o_Bc     = o_B2     + 409600;       // [128,4096] bf16
    const size_t o_Wrb    = o_Bc     + 1048576;      // [1024,160,32] bf16
    const size_t o_sA     = o_Wrb    + 10485760;     // im2col scratch
    const size_t o_sC     = o_sA     + 73932800;     // conv1 chunk out

    bf16*  pooled  = (bf16*)(ws + o_pooled);
    bf16*  conv2o  = (bf16*)(ws + o_conv2);
    float* capsO   = (float*)(ws + o_caps);
    bf16*  uBuf    = (bf16*)(ws + o_u);
    float* uhat    = (float*)(ws + o_uhat);
    float* blog    = (float*)(ws + o_blog);
    bf16*  B1t     = (bf16*)(ws + o_B1);
    bf16*  B2t     = (bf16*)(ws + o_B2);
    bf16*  Bct     = (bf16*)(ws + o_Bc);
    bf16*  Wrb     = (bf16*)(ws + o_Wrb);
    bf16*  sA      = (bf16*)(ws + o_sA);
    bf16*  sC      = (bf16*)(ws + o_sC);

    prepB1t<<<(128*96 + 255)/256, 256, 0, stream>>>(w1, B1t);
    prepB2t<<<(64*3200 + 255)/256, 256, 0, stream>>>(w2, B2t);
    prepBct<<<(128*4096 + 255)/256, 256, 0, stream>>>(wc, Bct);
    prepWrBf<<<(int)((1024UL*160*32 + 255)/256), 256, 0, stream>>>(Wr, Wrb);
    zero_f32<<<(655360 + 255)/256, 256, 0, stream>>>(blog, 655360);

    // conv1 (M=56448,N=128,K=96) + ReLU, maxpool, chunked over batch
    for (int ch = 0; ch < 8; ++ch) {
        const int b0 = ch * 8;
        im2col1<<<(int)(((size_t)8*7056*96 + 255)/256), 256, 0, stream>>>(x, sA, b0);
        gemm_wmma_bf16<<<dim3(2, 441), 256, 0, stream>>>(sA, B1t, sC, b1,
                                                         56448, 128, 96, /*bf16+relu*/2);
        maxpool2<<<(int)(((size_t)8*42*42*128 + 255)/256), 256, 0, stream>>>(sC, pooled, b0);
    }

    // conv2 (M=11552,N=64,K=3200), chunked
    for (int ch = 0; ch < 8; ++ch) {
        const int b0 = ch * 8;
        im2col2<<<(int)(((size_t)8*1444*3200 + 255)/256), 256, 0, stream>>>(pooled, sA, b0);
        gemm_wmma_bf16<<<dim3(1, 91), 256, 0, stream>>>(sA, B2t,
            conv2o + (size_t)ch * 11552 * 64, b2, 11552, 64, 3200, /*bf16*/1);
    }

    // primary caps conv (M=2048,N=128,K=4096, stride2), chunked
    for (int ch = 0; ch < 8; ++ch) {
        const int b0 = ch * 8;
        im2col3<<<(int)(((size_t)8*256*4096 + 255)/256), 256, 0, stream>>>(conv2o, sA, b0);
        gemm_wmma_bf16<<<dim3(2, 16), 256, 0, stream>>>(sA, Bct,
            capsO + (size_t)ch * 2048 * 128, bc, 2048, 128, 4096, /*f32*/0);
    }

    squash_caps<<<8192, 256, 0, stream>>>(capsO, uBuf);
    uhat_wmma<<<1024, 256, 0, stream>>>(uBuf, Wrb, uhat);

    routing_iter<<<64, 256, 0, stream>>>(uhat, blog, out, 0);
    routing_iter<<<64, 256, 0, stream>>>(uhat, blog, out, 0);
    routing_iter<<<64, 256, 0, stream>>>(uhat, blog, out, 1);
}
